// VisionMambaClassifier_8624294330966
// MI455X (gfx1250) — compile-verified, matrix operand order checked by reading, not run on hardware
//
#include <hip/hip_runtime.h>

typedef unsigned short u16;
typedef __attribute__((ext_vector_type(16))) __bf16 v16bf;
typedef __attribute__((ext_vector_type(8)))  float  v8f;

#define B_  32
#define D_  256
#define L_  4096
#define NL_ 4
#define NC_ 1000
#define K_  256

__device__ __forceinline__ u16 f2bf(float f) {
  union { float f; unsigned u; } v; v.f = f;
  unsigned r = (v.u + 0x7FFFu + ((v.u >> 16) & 1u)) >> 16;
  return (u16)r;
}
__device__ __forceinline__ float bf2f(u16 h) {
  union { unsigned u; float f; } v; v.u = ((unsigned)h) << 16;
  return v.f;
}

// CDNA5 async global->LDS copy (no VGPR staging, tracked by ASYNCcnt).
__device__ __forceinline__ void async_g2l_b128(const void* g, void* l) {
  unsigned loff = (unsigned)(size_t)l;
  asm volatile("global_load_async_to_lds_b128 %0, %1, off"
               :: "v"(loff), "v"(g) : "memory");
}
__device__ __forceinline__ void wait_async0() {
  asm volatile("s_wait_asynccnt 0x0" ::: "memory");
}

// ---------------------------------------------------------------- cvt f32->bf16
__global__ void cvt_f32_bf16(const float* __restrict__ in, u16* __restrict__ out, int n) {
  int i = blockIdx.x * blockDim.x + threadIdx.x;
  if (i < n) out[i] = f2bf(in[i]);
}

// ---------------------------------------------------------------- patch embed
__global__ __launch_bounds__(256) void patch_embed(
    const float* __restrict__ x, const float* __restrict__ w,
    const float* __restrict__ bias, u16* __restrict__ y) {
  __shared__ float wS[256 * 48];
  __shared__ float xS[12][256];
  __shared__ float bS[256];
  const int b = blockIdx.x, hp = blockIdx.y, tid = threadIdx.x;
  for (int i = tid; i < 256 * 48; i += 256) wS[i] = w[i];
  bS[tid] = bias[tid];
  for (int i = tid; i < 12 * 256; i += 256) {
    int row = i >> 8, col = i & 255;
    int c = row >> 2, r = row & 3;
    xS[row][col] = x[(((size_t)b * 3 + c) * 256 + (hp * 4 + r)) * 256 + col];
  }
  __syncthreads();
  const int d = tid;
  const float* wd = &wS[d * 48];
  for (int wp = 0; wp < 64; ++wp) {
    float acc = bS[d];
#pragma unroll
    for (int c = 0; c < 3; ++c)
#pragma unroll
      for (int i = 0; i < 4; ++i)
#pragma unroll
        for (int j = 0; j < 4; ++j)
          acc += wd[c * 16 + i * 4 + j] * xS[c * 4 + i][wp * 4 + j];
    y[((size_t)b * L_ + hp * 64 + wp) * D_ + d] = f2bf(acc);
  }
}

// ---------------------------------------------------------------- bf16 WMMA GEMM
// C[M,256] = act(A[M,256] * W[256,256]^T + bias). 256 threads = 8 waves.
// Block tile 128x256 (full N): A streamed from HBM exactly once.
// Double-buffered LDS stages fed by CDNA5 async global->LDS copies so the
// copy of K-slice k+1 overlaps the WMMA work of slice k (ASYNCcnt pipeline).
// B fragments are register double-buffered to keep 2 LDS loads in flight.
#define BM 128
#define BK 32
#define LDT (BK + 8)  // padded LDS row (ushorts): 80B stride, 16B aligned

__global__ __launch_bounds__(256) void gemm_bf16(
    const u16* __restrict__ A, const u16* __restrict__ W,
    const float* __restrict__ bias, u16* __restrict__ C, int act) {
  __shared__ __align__(16) u16 As[2][BM][LDT];  // 20 KB
  __shared__ __align__(16) u16 Ws[2][D_][LDT];  // 40 KB
  __shared__ float bSh[D_];

  const int tid = threadIdx.x;
  const int wave = tid >> 5, lane = tid & 31;
  const int mBase = blockIdx.x * BM;

  bSh[tid] = bias[tid];

  v8f acc[16];
#pragma unroll
  for (int i = 0; i < 16; ++i) acc[i] = (v8f)0.0f;

  const int aRow  = tid >> 1;         // 0..127
  const int aHalf = (tid & 1) * 16;   // 0 / 16 ushorts

  // stage K-slice kk into LDS buffer s (A: 32B/thread, W: 64B row/thread)
  auto stage_tiles = [&](int kk, int s) {
    const u16* gA = &A[(size_t)(mBase + aRow) * K_ + kk + aHalf];
    async_g2l_b128(gA,     &As[s][aRow][aHalf]);
    async_g2l_b128(gA + 8, &As[s][aRow][aHalf + 8]);
    const u16* gW = &W[(size_t)tid * K_ + kk];
#pragma unroll
    for (int i = 0; i < 4; ++i)
      async_g2l_b128(gW + i * 8, &Ws[s][tid][i * 8]);
  };

  union Frag { v16bf v; uint4 q[2]; };
  const int fm = lane & 15, fk = (lane >> 4) * 16;

  stage_tiles(0, 0);
  int cur = 0;
  for (int kk = 0; kk < K_; kk += BK) {
    wait_async0();        // current stage's copies (this wave) complete
    __syncthreads();      // -> all waves' copies visible
    if (kk + BK < K_) stage_tiles(kk + BK, cur ^ 1);  // prefetch next stage

    Frag fa, fbA, fbB;
    const uint4* ap = (const uint4*)&As[cur][wave * 16 + fm][fk];
    fa.q[0] = ap[0]; fa.q[1] = ap[1];
    const uint4* bp0 = (const uint4*)&Ws[cur][fm][fk];
    fbA.q[0] = bp0[0]; fbA.q[1] = bp0[1];

#pragma unroll
    for (int nt = 0; nt < 16; ++nt) {
      Frag* curF = (nt & 1) ? &fbB : &fbA;
      Frag* nxtF = (nt & 1) ? &fbA : &fbB;
      if (nt + 1 < 16) {
        const uint4* bp = (const uint4*)&Ws[cur][(nt + 1) * 16 + fm][fk];
        nxtF->q[0] = bp[0]; nxtF->q[1] = bp[1];
      }
      acc[nt] = __builtin_amdgcn_wmma_f32_16x16x32_bf16(
          false, fa.v, false, curF->v, (short)0, acc[nt], false, false);
    }
    __syncthreads();  // all reads of stage `cur` done before it is re-filled
    cur ^= 1;
  }

  // epilogue: C/D layout -> lane (L%16)=col, VGPR r + 8*(L/16)=row
  const int rowHi = (lane >> 4) * 8;
  const int colIn = lane & 15;
#pragma unroll
  for (int nt = 0; nt < 16; ++nt) {
    const int n = nt * 16 + colIn;
    const float bv = bSh[n];
#pragma unroll
    for (int r = 0; r < 8; ++r) {
      const int m = mBase + wave * 16 + rowHi + r;
      float v = acc[nt][r] + bv;
      if (act) v = 1.0f / (1.0f + __expf(-v));
      C[(size_t)m * D_ + n] = f2bf(v);
    }
  }
}

// ---------------------------------------------------------------- chunked linear scan
// h_t = (1-g_t) h_{t-1} + g_t u_t == affine h -> a*h + b. 64 chunks of 64 steps.
// 128 threads handle 2 packed bf16 lanes each (4B loads).
#define NCH 64
#define CT  (L_ / NCH)

__global__ __launch_bounds__(128) void scan_p1(
    const u16* __restrict__ G, const u16* __restrict__ U,
    float* __restrict__ cA, float* __restrict__ cB) {
  const int b = blockIdx.x, c = blockIdx.y, t2 = threadIdx.x;
  const unsigned* Gp = (const unsigned*)G;
  const unsigned* Up = (const unsigned*)U;
  size_t base = (((size_t)b * L_ + (size_t)c * CT) * D_ >> 1) + t2;
  float A0 = 1.f, B0 = 0.f, A1 = 1.f, B1 = 0.f;
  for (int t = 0; t < CT; ++t) {
    unsigned gv = Gp[base + (size_t)t * (D_ / 2)];
    unsigned uv = Up[base + (size_t)t * (D_ / 2)];
    float g0 = bf2f((u16)gv), g1 = bf2f((u16)(gv >> 16));
    float u0 = bf2f((u16)uv), u1 = bf2f((u16)(uv >> 16));
    float a0 = 1.f - g0, a1 = 1.f - g1;
    B0 = a0 * B0 + g0 * u0; A0 *= a0;
    B1 = a1 * B1 + g1 * u1; A1 *= a1;
  }
  size_t idx = ((size_t)b * NCH + c) * D_ + 2 * t2;
  cA[idx] = A0; cA[idx + 1] = A1;
  cB[idx] = B0; cB[idx + 1] = B1;
}

__global__ __launch_bounds__(256) void scan_p2(
    const float* __restrict__ cA, const float* __restrict__ cB,
    float* __restrict__ carry) {
  const int b = blockIdx.x, d = threadIdx.x;
  float h = 0.f;
  for (int c = 0; c < NCH; ++c) {
    size_t idx = ((size_t)b * NCH + c) * D_ + d;
    carry[idx] = h;
    h = cA[idx] * h + cB[idx];
  }
}

__global__ __launch_bounds__(128) void scan_p3(
    u16* __restrict__ G, const u16* __restrict__ U,
    const float* __restrict__ carry) {
  const int b = blockIdx.x, c = blockIdx.y, t2 = threadIdx.x;
  unsigned* Gp = (unsigned*)G;
  const unsigned* Up = (const unsigned*)U;
  size_t base = (((size_t)b * L_ + (size_t)c * CT) * D_ >> 1) + t2;
  size_t cidx = ((size_t)b * NCH + c) * D_ + 2 * t2;
  float h0 = carry[cidx], h1 = carry[cidx + 1];
  for (int t = 0; t < CT; ++t) {
    size_t idx = base + (size_t)t * (D_ / 2);
    unsigned gv = Gp[idx], uv = Up[idx];
    float g0 = bf2f((u16)gv), g1 = bf2f((u16)(gv >> 16));
    float u0 = bf2f((u16)uv), u1 = bf2f((u16)(uv >> 16));
    h0 = (1.f - g0) * h0 + g0 * u0;
    h1 = (1.f - g1) * h1 + g1 * u1;
    Gp[idx] = (unsigned)f2bf(h0) | ((unsigned)f2bf(h1) << 16);
  }
}

// ---------------------------------------------------------------- two-phase pool
__global__ __launch_bounds__(128) void pool_p1(const u16* __restrict__ X,
                                               float* __restrict__ part) {
  const int b = blockIdx.x, c = blockIdx.y, t2 = threadIdx.x;
  const unsigned* Xp = (const unsigned*)X;
  size_t base = (((size_t)b * L_ + (size_t)c * CT) * D_ >> 1) + t2;
  float s0 = 0.f, s1 = 0.f;
  for (int t = 0; t < CT; ++t) {
    unsigned v = Xp[base + (size_t)t * (D_ / 2)];
    s0 += bf2f((u16)v);
    s1 += bf2f((u16)(v >> 16));
  }
  size_t idx = ((size_t)b * NCH + c) * D_ + 2 * t2;
  part[idx] = s0; part[idx + 1] = s1;
}

__global__ __launch_bounds__(256) void pool_p2(const float* __restrict__ part,
                                               float* __restrict__ P) {
  const int b = blockIdx.x, d = threadIdx.x;
  float s = 0.f;
  for (int c = 0; c < NCH; ++c) s += part[((size_t)b * NCH + c) * D_ + d];
  P[b * D_ + d] = s * (1.0f / L_);
}

// ---------------------------------------------------------------- classifier
__global__ __launch_bounds__(256) void cls_k(
    const float* __restrict__ P, const float* __restrict__ Wc,
    const float* __restrict__ bc, float* __restrict__ out) {
  __shared__ float pS[D_];
  const int b = blockIdx.x;
  const int n = blockIdx.y * 256 + threadIdx.x;
  pS[threadIdx.x] = P[b * D_ + threadIdx.x];
  __syncthreads();
  if (n < NC_) {
    float acc = bc[n];
    for (int d2 = 0; d2 < D_; ++d2) acc += pS[d2] * Wc[(size_t)n * D_ + d2];
    out[(size_t)b * NC_ + n] = acc;
  }
}

// ---------------------------------------------------------------- driver
extern "C" void kernel_launch(void* const* d_in, const int* in_sizes, int n_in,
                              void* d_out, int out_size, void* d_ws, size_t ws_size,
                              hipStream_t stream) {
  const float* x      = (const float*)d_in[0];
  const float* conv_w = (const float*)d_in[1];
  const float* conv_b = (const float*)d_in[2];
  const float* gate_w = (const float*)d_in[3];
  const float* gate_b = (const float*)d_in[4];
  const float* out_w  = (const float*)d_in[5];
  const float* out_b  = (const float*)d_in[6];
  const float* cls_w  = (const float*)d_in[7];
  const float* cls_b  = (const float*)d_in[8];

  char* ws = (char*)d_ws;
  size_t off = 0;
  const size_t actElems = (size_t)B_ * L_ * D_;  // 33,554,432
  u16* bufX = (u16*)(ws + off); off += actElems * sizeof(u16);  // 64 MB
  u16* bufG = (u16*)(ws + off); off += actElems * sizeof(u16);  // 64 MB
  u16* wG   = (u16*)(ws + off); off += (size_t)NL_ * D_ * D_ * sizeof(u16);
  u16* wO   = (u16*)(ws + off); off += (size_t)NL_ * D_ * D_ * sizeof(u16);
  float* cA    = (float*)(ws + off); off += (size_t)B_ * NCH * D_ * sizeof(float);
  float* cB    = (float*)(ws + off); off += (size_t)B_ * NCH * D_ * sizeof(float);
  float* carry = (float*)(ws + off); off += (size_t)B_ * NCH * D_ * sizeof(float);
  float* pooled = (float*)(ws + off); off += (size_t)B_ * D_ * sizeof(float);

  {
    int n = NL_ * D_ * D_;
    cvt_f32_bf16<<<dim3((n + 255) / 256), dim3(256), 0, stream>>>(gate_w, wG, n);
    cvt_f32_bf16<<<dim3((n + 255) / 256), dim3(256), 0, stream>>>(out_w, wO, n);
  }

  patch_embed<<<dim3(B_, 64), dim3(256), 0, stream>>>(x, conv_w, conv_b, bufX);

  const dim3 gemmGrid((B_ * L_) / BM);  // 1024 blocks, full-N strip each
  for (int i = 0; i < NL_; ++i) {
    gemm_bf16<<<gemmGrid, dim3(256), 0, stream>>>(
        bufX, wG + (size_t)i * D_ * D_, gate_b + (size_t)i * D_, bufG, 1);
    scan_p1<<<dim3(B_, NCH), dim3(128), 0, stream>>>(bufG, bufX, cA, cB);
    scan_p2<<<dim3(B_), dim3(256), 0, stream>>>(cA, cB, carry);
    scan_p3<<<dim3(B_, NCH), dim3(128), 0, stream>>>(bufG, bufX, carry);
    gemm_bf16<<<gemmGrid, dim3(256), 0, stream>>>(
        bufG, wO + (size_t)i * D_ * D_, out_b + (size_t)i * D_, bufX, 0);
  }

  pool_p1<<<dim3(B_, NCH), dim3(128), 0, stream>>>(bufX, cA);
  pool_p2<<<dim3(B_), dim3(256), 0, stream>>>(cA, pooled);
  cls_k<<<dim3(B_, (NC_ + 255) / 256), dim3(256), 0, stream>>>(
      pooled, cls_w, cls_b, (float*)d_out);
}
